// InverseTransform2D_23313082482908
// MI455X (gfx1250) — matrix operand order repaired and checked by reading, not compiled, not run on hardware
//
#include <hip/hip_runtime.h>
#include <math.h>

typedef __attribute__((ext_vector_type(16))) _Float16 v16h;
typedef __attribute__((ext_vector_type(8)))  float    v8f;
typedef __attribute__((ext_vector_type(4)))  float    f32x4;

// ---------------- workspace layout (floats) ----------------
#define OFF_HACC 0          // 32*1008 f32 GEMM1 accumulator (split-K atomics)
#define OFF_H2   32256      // 32*32   f32 relu(layer2) output
#define OFF_SAIT 33280      // 32      sum ai_t per batch
#define OFF_QAIT 33312      // 32      sum ai_t^2
#define OFF_ST   33344      // 32      sum t
#define OFF_QT   33376      // 32      sum t^2
#define OFF_STS  33408      // 32      sum ts
#define OFF_QTS  33440      // 32      sum ts^2
#define OFF_SC   33472      // 2       [0]=sum (ai_x-ai_t)^2, [1]=sum (xs-ts)^2
#define WS_FLOATS 33474

#define HW 224
#define NPIX 50176          // 224*224
#define KDIM 100352         // 2*NPIX
#define NCOLS 1000
#define NPAD 1008           // 63 tiles * 16

__device__ __forceinline__ float clip01(float v) { return fminf(fmaxf(v, 0.0f), 1.0f); }

// ---------------- zero workspace ----------------
__global__ void zero_ws_kernel(float* ws, int n) {
    int i = blockIdx.x * blockDim.x + threadIdx.x;
    if (i < n) ws[i] = 0.0f;
}

// ---------------- stage 1: pixel losses + reductions ----------------
// dist collapses to: 0.1 if any pixel >0 within Chebyshev radius 9, else 0.
// Edge-clamped halo is exactly equivalent for both the dilation test and the
// replicate-padded 3x3 erosion (clamp projects into the clipped window).
__global__ __launch_bounds__(256) void stage1_kernel(
    const float* __restrict__ inputs, const float* __restrict__ targets, float* ws)
{
    __shared__ float sx[50][52];
    __shared__ float stt[50][52];
    __shared__ float rmx[50][32];
    __shared__ float rmt[50][32];
    __shared__ float red[256];

    const int b   = blockIdx.z;
    const int gr0 = blockIdx.y * 32;
    const int gc0 = blockIdx.x * 32;
    const int tid = threadIdx.x;
    const float* inb = inputs  + (size_t)b * NPIX;
    const float* tgb = targets + (size_t)b * NPIX;

    // load 50x50 clipped tile with edge clamp
    for (int idx = tid; idx < 2500; idx += 256) {
        int r = idx / 50, c = idx % 50;
        int gr = gr0 + r - 9; gr = gr < 0 ? 0 : (gr > 223 ? 223 : gr);
        int gc = gc0 + c - 9; gc = gc < 0 ? 0 : (gc > 223 ? 223 : gc);
        sx[r][c]  = clip01(inb[gr * HW + gc]);
        stt[r][c] = clip01(tgb[gr * HW + gc]);
    }
    __syncthreads();

    // separable 19-wide row max
    for (int idx = tid; idx < 1600; idx += 256) {
        int r = idx / 32, j = idx % 32;
        float mx = -1.0f, mt = -1.0f;
#pragma unroll
        for (int d = 0; d < 19; ++d) {
            mx = fmaxf(mx, sx[r][j + d]);
            mt = fmaxf(mt, stt[r][j + d]);
        }
        rmx[r][j] = mx; rmt[r][j] = mt;
    }
    __syncthreads();

    const float E1 = 0.36787944117144233f; // exp(-1)
    float a0 = 0.f, a1 = 0.f, a2 = 0.f, a3 = 0.f, a4 = 0.f, a5 = 0.f, a6 = 0.f, a7 = 0.f;

    for (int idx = tid; idx < 1024; idx += 256) {
        int i = idx / 32, j = idx % 32;
        int r = i + 9, c = j + 9;
        float cmx = -1.0f, cmt = -1.0f;
#pragma unroll
        for (int d = 0; d < 19; ++d) {
            cmx = fmaxf(cmx, rmx[i + d][j]);
            cmt = fmaxf(cmt, rmt[i + d][j]);
        }
        float fx = (cmx > 0.0f) ? E1 : 1.0f;
        float ft = (cmt > 0.0f) ? E1 : 1.0f;
        float x = sx[r][c], t = stt[r][c];
        float alpha = 1.0f / (1.0f + __expf(-0.5f * t));
        float aix = alpha * fx, ait = alpha * ft;

        float mnx = 1e30f, mnt = 1e30f;
#pragma unroll
        for (int dr = -1; dr <= 1; ++dr)
#pragma unroll
            for (int dc = -1; dc <= 1; ++dc) {
                mnx = fminf(mnx, sx[r + dr][c + dc]);
                mnt = fminf(mnt, stt[r + dr][c + dc]);
            }
        float xs = x - mnx, tsv = t - mnt;

        float d0 = aix - ait; a0 += d0 * d0;
        a1 += ait; a2 += ait * ait;
        a3 += t;   a4 += t * t;
        float d1 = xs - tsv; a5 += d1 * d1;
        a6 += tsv; a7 += tsv * tsv;
    }
    __syncthreads();

    float accs[8] = {a0, a1, a2, a3, a4, a5, a6, a7};
    float* dst[8] = {ws + OFF_SC + 0, ws + OFF_SAIT + b, ws + OFF_QAIT + b,
                     ws + OFF_ST + b, ws + OFF_QT + b,   ws + OFF_SC + 1,
                     ws + OFF_STS + b, ws + OFF_QTS + b};
#pragma unroll
    for (int q = 0; q < 8; ++q) {
        red[tid] = accs[q];
        __syncthreads();
        for (int s = 128; s > 0; s >>= 1) {
            if (tid < s) red[tid] += red[tid + s];
            __syncthreads();
        }
        if (tid == 0) atomicAdd(dst[q], red[0]);
        __syncthreads();
    }
}

// ---------------- GEMM1 helpers: WMMA fragment loads ----------------
__device__ __forceinline__ v16h load_feat_a(const float* p) {
    // A 16x32 f16: elements 0..7 <-> K=off..off+7, 8..15 <-> K=16+off..16+off+7
    f32x4 u0 = *reinterpret_cast<const f32x4*>(p);
    f32x4 u1 = *reinterpret_cast<const f32x4*>(p + 4);
    f32x4 u2 = *reinterpret_cast<const f32x4*>(p + 16);
    f32x4 u3 = *reinterpret_cast<const f32x4*>(p + 20);
    v16h a;
#pragma unroll
    for (int e = 0; e < 4; ++e) {
        a[e]      = (_Float16)clip01(u0[e]);
        a[4 + e]  = (_Float16)clip01(u1[e]);
        a[8 + e]  = (_Float16)clip01(u2[e]);
        a[12 + e] = (_Float16)clip01(u3[e]);
    }
    return a;
}

__device__ __forceinline__ v16h load_w1_b(const float* p) {
    // B 32x16 f16: elements 0..7 <-> K=off..off+7, 8..15 <-> K=16+off..16+off+7
    // Non-temporal: w1 is a 401 MB single-use stream; keep feat L2-resident.
    f32x4 u0 = __builtin_nontemporal_load(reinterpret_cast<const f32x4*>(p));
    f32x4 u1 = __builtin_nontemporal_load(reinterpret_cast<const f32x4*>(p + 4));
    f32x4 u2 = __builtin_nontemporal_load(reinterpret_cast<const f32x4*>(p + 16));
    f32x4 u3 = __builtin_nontemporal_load(reinterpret_cast<const f32x4*>(p + 20));
    v16h bm;
#pragma unroll
    for (int e = 0; e < 4; ++e) {
        bm[e]      = (_Float16)u0[e];
        bm[4 + e]  = (_Float16)u1[e];
        bm[8 + e]  = (_Float16)u2[e];
        bm[12 + e] = (_Float16)u3[e];
    }
    return bm;
}

// ---------------- GEMM1: feat[32x100352] @ w1^T -> h_acc[32x1008] ----------------
// grid(63, 14), 256 threads = 8 waves; wave handles 28 K-steps of 32.
// w1 (401 MB) streamed exactly once from HBM (non-temporal); feat re-reads hit L2.
// Each wave's 896-wide K chunk never straddles the inputs/targets boundary
// (50176 = 56*896), so the source select is hoisted out of the loop and made
// scalar via readfirstlane (k0 is wave-uniform).
__global__ __launch_bounds__(256) void gemm1_kernel(
    const float* __restrict__ inputs, const float* __restrict__ targets,
    const float* __restrict__ w1, float* __restrict__ h_acc)
{
    const int tile   = blockIdx.x;           // N tile: cols [tile*16, tile*16+16)
    const int kchunk = blockIdx.y;           // 0..13
    const int tid  = threadIdx.x;
    const int wave = tid >> 5;
    const int lane = tid & 31;

    const int colbase = tile * 16;
    int col = colbase + (lane & 15);
    int colc = col > 999 ? 999 : col;        // clamp: padded cols produce unused garbage

    const int aoff  = (lane < 16) ? 0 : 8;   // A fragment K offset
    const int koffB = (lane < 16) ? 0 : 8;   // B fragment K offset
    const int b0 = lane & 15;
    const int b1 = b0 + 16;

    // wave-uniform start of this wave's K range (scalar)
    const int k0 = __builtin_amdgcn_readfirstlane((kchunk * 224 + wave * 28) * 32);

    // hoisted source select: whole chunk is on one side of the NPIX boundary
    const float* abase;
    int akoff;
    if (k0 < NPIX) { abase = inputs;  akoff = k0; }
    else           { abase = targets; akoff = k0 - NPIX; }
    const float* pa0 = abase + (size_t)b0 * NPIX + akoff + aoff;
    const float* pa1 = abase + (size_t)b1 * NPIX + akoff + aoff;
    const float* pb  = w1 + (size_t)colc * KDIM + k0 + koffB;

    v8f c0 = {}; v8f c1 = {};
#pragma unroll 4
    for (int s = 0; s < 28; ++s) {
        v16h amat0 = load_feat_a(pa0 + s * 32);
        v16h amat1 = load_feat_a(pa1 + s * 32);
        v16h bmat  = load_w1_b(pb + s * 32);

        c0 = __builtin_amdgcn_wmma_f32_16x16x32_f16(false, amat0, false, bmat,
                                                    (short)0, c0, false, false);
        c1 = __builtin_amdgcn_wmma_f32_16x16x32_f16(false, amat1, false, bmat,
                                                    (short)0, c1, false, false);
    }

    // C layout: element r <-> row (r + 8*(lane>=16)), col lane&15
    const int nc = colbase + (lane & 15);
    const int rbase = (lane >= 16) ? 8 : 0;
#pragma unroll
    for (int r = 0; r < 8; ++r) {
        atomicAdd(&h_acc[(size_t)(r + rbase) * NPAD + nc],       c0[r]);
        atomicAdd(&h_acc[(size_t)(16 + r + rbase) * NPAD + nc],  c1[r]);
    }
}

// ---------------- GEMM2: relu(h+b1) @ w2^T -> relu(..+b2) [32x32] ----------------
__global__ __launch_bounds__(1024) void gemm2_kernel(
    const float* __restrict__ h_acc, const float* __restrict__ b1,
    const float* __restrict__ w2, const float* __restrict__ b2,
    float* __restrict__ h2)
{
    __shared__ float hs[32][65];
    __shared__ float w2s[32][65];
    const int bidx = threadIdx.x;   // batch
    const int j2   = threadIdx.y;   // out feature
    const int tid  = j2 * 32 + bidx;

    float s = b2[j2];
    for (int kb = 0; kb < NCOLS; kb += 64) {
        for (int idx = tid; idx < 2048; idx += 1024) {
            int row = idx >> 6, kk = idx & 63;
            int k = kb + kk;
            float hv = 0.f, wv = 0.f;
            if (k < NCOLS) {
                hv = fmaxf(h_acc[(size_t)row * NPAD + k] + b1[k], 0.0f);
                wv = w2[(size_t)row * NCOLS + k];
            }
            hs[row][kk]  = hv;
            w2s[row][kk] = wv;
        }
        __syncthreads();
#pragma unroll 8
        for (int kk = 0; kk < 64; ++kk)
            s += hs[bidx][kk] * w2s[j2][kk];
        __syncthreads();
    }
    h2[(size_t)bidx * 32 + j2] = fmaxf(s, 0.0f);
}

// ---------------- final: layer3, norms, loss assembly ----------------
__global__ __launch_bounds__(128) void final_kernel(
    const float* __restrict__ ws, const float* __restrict__ w3,
    const float* __restrict__ b3, float* __restrict__ out)
{
    __shared__ float cf[32][4];
    __shared__ float nrm[32];
    const float* h2 = ws + OFF_H2;
    int tid = threadIdx.x;
    {
        int b = tid >> 2, j = tid & 3;
        float s = b3[j];
#pragma unroll
        for (int k = 0; k < 32; ++k) s += h2[b * 32 + k] * w3[j * 32 + k];
        cf[b][j] = s;
    }
    __syncthreads();
    if (tid < 32) {
        float s = 0.f;
#pragma unroll
        for (int j = 0; j < 4; ++j) s += cf[tid][j] * cf[tid][j];
        nrm[tid] = sqrtf(s);
    }
    __syncthreads();
    if (tid == 0) {
        const float M = 1605632.0f;       // 32*224*224
        const float C = 50174.0f;         // N-2
        float base = 0.f, lr_ait = 0.f, lr_t = 0.f, lr_ts = 0.f;
        for (int b = 0; b < 32; ++b) {
            base += nrm[b];
            float S;
            S = ws[OFF_SAIT + b]; lr_ait += C * S * S + ws[OFF_QAIT + b];
            S = ws[OFF_ST + b];   lr_t   += C * S * S + ws[OFF_QT + b];
            S = ws[OFF_STS + b];  lr_ts  += C * S * S + ws[OFF_QTS + b];
        }
        base *= (1.0f / 32.0f);
        float dt_loss   = ws[OFF_SC + 0] / M + 0.3f * lr_ait / M;
        float skel_loss = ws[OFF_SC + 1] / M + 0.3f * lr_ts / M;
        float inv_loss  = base + 0.3f * lr_t / M;
        out[0] = 0.3f * inv_loss + 0.2f * dt_loss + 0.2f * skel_loss;
    }
}

extern "C" void kernel_launch(void* const* d_in, const int* in_sizes, int n_in,
                              void* d_out, int out_size, void* d_ws, size_t ws_size,
                              hipStream_t stream) {
    const float* inputs  = (const float*)d_in[0];
    const float* targets = (const float*)d_in[1];
    const float* w1 = (const float*)d_in[2];
    const float* b1 = (const float*)d_in[3];
    const float* w2 = (const float*)d_in[4];
    const float* b2 = (const float*)d_in[5];
    const float* w3 = (const float*)d_in[6];
    const float* b3 = (const float*)d_in[7];
    float* ws  = (float*)d_ws;
    float* out = (float*)d_out;

    zero_ws_kernel<<<(WS_FLOATS + 255) / 256, 256, 0, stream>>>(ws, WS_FLOATS);
    stage1_kernel<<<dim3(7, 7, 32), 256, 0, stream>>>(inputs, targets, ws);
    gemm1_kernel<<<dim3(63, 14), 256, 0, stream>>>(inputs, targets, w1, ws + OFF_HACC);
    gemm2_kernel<<<1, dim3(32, 32), 0, stream>>>(ws + OFF_HACC, b1, w2, b2, ws + OFF_H2);
    final_kernel<<<1, 128, 0, stream>>>(ws, w3, b3, out);
}
